// BipartiteGIN_55825984913938
// MI455X (gfx1250) — matrix operand order repaired and checked by reading, not compiled, not run on hardware
//
#include <hip/hip_runtime.h>

// BipartiteGIN on MI455X (gfx1250, wave32, WMMA).
// All K=128 GEMMs -> v_wmma_f32_16x16x32_bf16 (bf16 operands, f32 accum).
// Edge-MLP layer1 generated on the fly inside the GEMM A-tile loader.
// BatchNorm stats fused into GEMM epilogue; BN-apply fused into scatter.
// Round 3: full-tile fast path with b128 global loads + b64 LDS stores
// (kills the per-element saveexec guards); 4-wide elementwise kernels.

#define DH   128
#define NSN  50000
#define NTN  50000
#define EN   600000
#define BN_EPS 1e-5f

typedef __attribute__((ext_vector_type(16))) __bf16 v16bf;
typedef __attribute__((ext_vector_type(8)))  float  v8f;

union FragAB { v16bf v; unsigned u[8]; };
union FragC  { v8f   v; float    f[8]; };

__device__ __forceinline__ unsigned short f2bf(float f) {
  unsigned u = __float_as_uint(f);
  unsigned r = u + 0x7FFFu + ((u >> 16) & 1u);   // round-to-nearest-even
  return (unsigned short)(r >> 16);
}
__device__ __forceinline__ unsigned pack2bf(float a, float b) {
  return (unsigned)f2bf(a) | ((unsigned)f2bf(b) << 16);
}
__device__ __forceinline__ float bf2f(unsigned short h) {
  return __uint_as_float(((unsigned)h) << 16);
}

// ---------------------------------------------------------------------------
// Tiled GEMM: Y[M,128] = act( A[M,128] @ W[128,128] + bias (+ rs[m]*rw[n]) )
// AMODE 0: A is a strided f32 matrix (lda).
// AMODE 1: A row r is relu(edge_attr[r]*eW1 + eB1)  (edge-MLP layer 1, fused).
// STATS: accumulate column sum / sum-of-squares (for BatchNorm) via LDS + atomics.
// OUTBF16: store output as bf16 (edge activation buffer).
// Block = 256 threads = 8 wave32; block tile 128x128; wave tile 16x128.
// ---------------------------------------------------------------------------
template<int AMODE, bool RELU, bool RANK1, bool STATS, bool OUTBF16>
__global__ __launch_bounds__(256) void gemm128_wmma(
    int M,
    const float* __restrict__ A, int lda,
    const float* __restrict__ eattr,
    const float* __restrict__ eW1, const float* __restrict__ eB1,
    const float* __restrict__ W,   const float* __restrict__ bias,
    const float* __restrict__ rs,  const float* __restrict__ rw,
    float* __restrict__ C, int ldc,
    unsigned short* __restrict__ Cb,
    float* __restrict__ ssum, float* __restrict__ ssq)
{
  __shared__ unsigned short ldsA[128][36];  // [m][k] bf16; pad->72B row, b64-aligned
  __shared__ unsigned short ldsB[128][36];  // [n][k] bf16 (k-major per column)
  __shared__ float ls_sum[128];
  __shared__ float ls_sq[128];

  const int tid  = threadIdx.x;
  const int wv   = tid >> 5;        // wave 0..7 -> rows [16*wv, 16*wv+16)
  const int lane = tid & 31;
  const int hi   = lane >> 4;
  const int ln   = lane & 15;
  const int row0 = blockIdx.x * 128;
  const bool full = (row0 + 128 <= M);   // all but the last block

  if (STATS && tid < 128) { ls_sum[tid] = 0.f; ls_sq[tid] = 0.f; }

  FragC acc[8];
  #pragma unroll
  for (int nt = 0; nt < 8; ++nt)
    #pragma unroll
    for (int j = 0; j < 8; ++j) acc[nt].f[j] = 0.f;

  for (int kt = 0; kt < 4; ++kt) {              // K = 128 = 4 x 32
    const int k0 = kt * 32;
    __syncthreads();
    // --- A tile: 128 rows x 32 k, f32 -> bf16 ---
    if (full) {
      // fast path: unguarded b128 loads, packed b64 LDS stores
      #pragma unroll
      for (int it = 0; it < 4; ++it) {
        int idx = tid + it * 256;          // 0..1023
        int m = idx >> 3, k4 = (idx & 7) * 4;
        float4 s;
        if (AMODE == 0) {
          s = *(const float4*)&A[(size_t)(row0 + m) * lda + k0 + k4];
        } else {
          float a = eattr[row0 + m];
          float4 w1 = *(const float4*)&eW1[k0 + k4];
          float4 b1 = *(const float4*)&eB1[k0 + k4];
          s.x = fmaxf(fmaf(a, w1.x, b1.x), 0.f);
          s.y = fmaxf(fmaf(a, w1.y, b1.y), 0.f);
          s.z = fmaxf(fmaf(a, w1.z, b1.z), 0.f);
          s.w = fmaxf(fmaf(a, w1.w, b1.w), 0.f);
        }
        uint2 p; p.x = pack2bf(s.x, s.y); p.y = pack2bf(s.z, s.w);
        *(uint2*)&ldsA[m][k4] = p;
      }
    } else {
      // partial last block: guarded scalar path (zero-pad rows >= M)
      #pragma unroll
      for (int it = 0; it < 16; ++it) {
        int idx = tid + it * 256;
        int m = idx >> 5, k = idx & 31;
        int row = row0 + m;
        float v = 0.f;
        if (row < M) {
          if (AMODE == 0) v = A[(size_t)row * lda + k0 + k];
          else {
            float a = eattr[row];
            v = fmaxf(fmaf(a, eW1[k0 + k], eB1[k0 + k]), 0.f);
          }
        }
        ldsA[m][k] = f2bf(v);
      }
    }
    // --- B tile: W[k0+k][n4..n4+3] b128 loads, transposed b16 LDS stores ---
    #pragma unroll
    for (int it = 0; it < 4; ++it) {
      int idx = tid + it * 256;            // 0..1023
      int k = idx >> 5, n4 = (idx & 31) * 4;
      float4 w = *(const float4*)&W[(size_t)(k0 + k) * 128 + n4];
      ldsB[n4 + 0][k] = f2bf(w.x);
      ldsB[n4 + 1][k] = f2bf(w.y);
      ldsB[n4 + 2][k] = f2bf(w.z);
      ldsB[n4 + 3][k] = f2bf(w.w);
    }
    __syncthreads();
    // --- build A fragment (ISA 7.12.2: 16-bit A 16x32 layout) ---
    FragAB fa;
    #pragma unroll
    for (int p = 0; p < 8; ++p) {
      int kb = (p < 4 ? 2 * p : 16 + 2 * (p - 4)) + 8 * hi;  // even -> dword read
      fa.u[p] = *(const unsigned*)&ldsA[wv * 16 + ln][kb];
    }
    // --- preload ALL 8 B fragments (64 VGPRs) so WMMAs issue back-to-back ---
    FragAB fb[8];
    #pragma unroll
    for (int nt = 0; nt < 8; ++nt)
      #pragma unroll
      for (int p = 0; p < 8; ++p)
        fb[nt].u[p] = *(const unsigned*)&ldsB[nt * 16 + ln][(p < 4 ? 2 * p : 16 + 2 * (p - 4)) + 8 * hi];
    // --- 8 back-to-back WMMAs (independent accumulators, shared A) ---
    #pragma unroll
    for (int nt = 0; nt < 8; ++nt)
      acc[nt].v = __builtin_amdgcn_wmma_f32_16x16x32_bf16(
          false, fa.v, false, fb[nt].v, (short)0, acc[nt].v, false, false);
  }

  // --- epilogue: bias (+rank1) (+relu), store, fused BN statistics ---
  float bcol[8], wcol[8];
  #pragma unroll
  for (int nt = 0; nt < 8; ++nt) {
    int c = nt * 16 + ln;
    bcol[nt] = bias[c];
    wcol[nt] = RANK1 ? rw[c] : 0.f;
  }
  float psum[8], psq[8];
  #pragma unroll
  for (int nt = 0; nt < 8; ++nt) { psum[nt] = 0.f; psq[nt] = 0.f; }

  #pragma unroll
  for (int r = 0; r < 8; ++r) {                 // C layout: M = r + 8*hi
    int row = row0 + wv * 16 + r + 8 * hi;
    if (row >= M) continue;
    float sv = RANK1 ? rs[row] : 0.f;
    #pragma unroll
    for (int nt = 0; nt < 8; ++nt) {
      float y = acc[nt].f[r] + bcol[nt];
      if (RANK1) y = fmaf(sv, wcol[nt], y);
      if (RELU)  y = fmaxf(y, 0.f);
      size_t o = (size_t)row * ldc + nt * 16 + ln;
      if (OUTBF16) Cb[o] = f2bf(y); else C[o] = y;
      if (STATS) { psum[nt] += y; psq[nt] = fmaf(y, y, psq[nt]); }
    }
  }
  if (STATS) {
    #pragma unroll
    for (int nt = 0; nt < 8; ++nt) {
      atomicAdd(&ls_sum[nt * 16 + ln], psum[nt]);
      atomicAdd(&ls_sq [nt * 16 + ln], psq [nt]);
    }
    __syncthreads();
    if (tid < 128) {
      atomicAdd(&ssum[tid], ls_sum[tid]);
      atomicAdd(&ssq [tid], ls_sq [tid]);
    }
  }
}

// --------------------------- small helper kernels ---------------------------
__global__ void zero_kernel(float* p, long long n) {
  for (long long i = (long long)blockIdx.x * blockDim.x + threadIdx.x; i < n;
       i += (long long)gridDim.x * blockDim.x) p[i] = 0.f;
}

// Y[M,128] = relu(X[M,K] @ W[K,128] + b), small K (encoder layer 1)
__global__ void lin_small_relu(const float* __restrict__ X, int M, int K,
                               const float* __restrict__ W, const float* __restrict__ b,
                               float* __restrict__ Y) {
  long long g = (long long)blockIdx.x * 256 + threadIdx.x;
  if (g >= (long long)M * 128) return;
  int m = (int)(g >> 7), n = (int)(g & 127);
  float acc = b[n];
  for (int k = 0; k < K; ++k) acc = fmaf(X[(size_t)m * K + k], W[k * 128 + n], acc);
  Y[g] = fmaxf(acc, 0.f);
}

// ass[m] = sigmoid(H[m,:] . W2 + b2), one wave32 per row
__global__ void ass2_kernel(const float* __restrict__ H, int M,
                            const float* __restrict__ W2, const float* __restrict__ b2,
                            float* __restrict__ out) {
  int wv = threadIdx.x >> 5, lane = threadIdx.x & 31;
  int row = blockIdx.x * 8 + wv;
  if (row >= M) return;
  float acc = 0.f;
  #pragma unroll
  for (int j = 0; j < 4; ++j) {
    int k = lane + 32 * j;
    acc = fmaf(H[(size_t)row * 128 + k], W2[k], acc);
  }
  for (int off = 16; off; off >>= 1) acc += __shfl_xor(acc, off, 32);
  if (lane == 0) out[row] = 1.f / (1.f + __expf(-(acc + b2[0])));
}

// (sum, sumsq, count, gamma, beta) -> per-column affine (scale, shift)
__global__ void bn_finalize(const float* __restrict__ sum, const float* __restrict__ sq,
                            float count, const float* __restrict__ g,
                            const float* __restrict__ beta,
                            float* __restrict__ scale, float* __restrict__ shift) {
  int n = threadIdx.x;                 // 128 threads
  float m = sum[n] / count;
  float v = sq[n] / count - m * m;     // biased variance (matches reference)
  float sc = g[n] * rsqrtf(v + BN_EPS);
  scale[n] = sc;
  shift[n] = beta[n] - sc * m;
}

// Y[m, ldy] = relu(X[m,128]*scale + shift), 4 columns per thread (b128 loads)
__global__ void bn_relu_kernel(const float* __restrict__ X, int M,
                               const float* __restrict__ sc, const float* __restrict__ sh,
                               float* __restrict__ Y, int ldy) {
  long long g = (long long)blockIdx.x * 256 + threadIdx.x;
  if (g >= (long long)M * 32) return;
  int m = (int)(g >> 5), n = (int)(g & 31) * 4;
  float4 x = *(const float4*)&X[(size_t)m * 128 + n];
  float4 a = *(const float4*)&sc[n];
  float4 b = *(const float4*)&sh[n];
  float4 y;
  y.x = fmaxf(fmaf(x.x, a.x, b.x), 0.f);
  y.y = fmaxf(fmaf(x.y, a.y, b.y), 0.f);
  y.z = fmaxf(fmaf(x.z, a.z, b.z), 0.f);
  y.w = fmaxf(fmaf(x.w, a.w, b.w), 0.f);
  *(float4*)&Y[(size_t)m * ldy + n] = y;
}

// gin input: Y[m,128] = (1+eps)*Xprev[m, ldx] + agg[m,128], 4-wide
__global__ void gin_kernel(const float* __restrict__ Xprev, int ldx,
                           const float* __restrict__ agg, const float* __restrict__ eps,
                           int M, float* __restrict__ Y) {
  long long g = (long long)blockIdx.x * 256 + threadIdx.x;
  if (g >= (long long)M * 32) return;
  int m = (int)(g >> 5), n = (int)(g & 31) * 4;
  float e = 1.f + eps[0];
  float4 x = *(const float4*)&Xprev[(size_t)m * ldx + n];
  float4 a = *(const float4*)&agg[(size_t)m * 128 + n];
  float4 y;
  y.x = fmaf(e, x.x, a.x); y.y = fmaf(e, x.y, a.y);
  y.z = fmaf(e, x.z, a.z); y.w = fmaf(e, x.w, a.w);
  *(float4*)&Y[(size_t)m * 128 + n] = y;
}

// msg = relu( BN1(S[src_idx]) + BN2(ee) ); atomic segment-sum into agg[dst_idx]
// 4 columns per thread: b128 gather on S, b64 on bf16 edge buffer.
__global__ void scatter_kernel(const float* __restrict__ S,
                               const unsigned short* __restrict__ ee,
                               const int* __restrict__ sidx, const int* __restrict__ didx,
                               long long E,
                               const float* __restrict__ sc1, const float* __restrict__ sh1,
                               const float* __restrict__ sc2, const float* __restrict__ sh2,
                               float* __restrict__ agg) {
  long long g = (long long)blockIdx.x * 256 + threadIdx.x;
  if (g >= E * 32) return;
  long long e = g >> 5;
  int n = (int)(g & 31) * 4;
  int si = sidx[e], di = didx[e];
  float4 s4 = *(const float4*)&S[(size_t)si * 128 + n];
  uint2 eu  = *(const uint2*)&ee[(size_t)e * 128 + n];
  float4 a1 = *(const float4*)&sc1[n];
  float4 b1 = *(const float4*)&sh1[n];
  float4 a2 = *(const float4*)&sc2[n];
  float4 b2 = *(const float4*)&sh2[n];
  float m0 = fmaxf(fmaf(s4.x, a1.x, b1.x) + fmaf(bf2f((unsigned short)(eu.x & 0xffff)), a2.x, b2.x), 0.f);
  float m1 = fmaxf(fmaf(s4.y, a1.y, b1.y) + fmaf(bf2f((unsigned short)(eu.x >> 16)),   a2.y, b2.y), 0.f);
  float m2 = fmaxf(fmaf(s4.z, a1.z, b1.z) + fmaf(bf2f((unsigned short)(eu.y & 0xffff)), a2.z, b2.z), 0.f);
  float m3 = fmaxf(fmaf(s4.w, a1.w, b1.w) + fmaf(bf2f((unsigned short)(eu.y >> 16)),   a2.w, b2.w), 0.f);
  float* dst = &agg[(size_t)di * 128 + n];
  atomicAdd(dst + 0, m0);
  atomicAdd(dst + 1, m1);
  atomicAdd(dst + 2, m2);
  atomicAdd(dst + 3, m3);
}

// Wsum = W[0:128,:] + W[128:256,:]   (concat([x,x]) @ W fold)
__global__ void wsum_kernel(const float* __restrict__ W, float* __restrict__ out) {
  int i = blockIdx.x * 256 + threadIdx.x;
  if (i < 16384) out[i] = W[i] + W[i + 16384];
}

// ------------------------------ host side ----------------------------------
struct Lin { const float *W, *b; };
struct Blk { const float *beta, *g; Lin l1, l2; };   // jax sorted keys: beta,g,l1,l2; W<b
struct VarL { Lin ass1, ass2; Blk edge; const float* eps; Blk joint; Blk mlp; };
struct ConL { Blk edge; const float* eps; Blk joint; Blk mlp; };

static inline const float* nxt(void* const* d, int& c) { return (const float*)d[c++]; }
static inline Lin nxtLin(void* const* d, int& c) { Lin l; l.W = nxt(d,c); l.b = nxt(d,c); return l; }
static inline Blk nxtBlk(void* const* d, int& c) {
  Blk b; b.beta = nxt(d,c); b.g = nxt(d,c); b.l1 = nxtLin(d,c); b.l2 = nxtLin(d,c); return b;
}

#define GRID_EW(n) dim3((unsigned)(((long long)(n) + 255) / 256)), dim3(256)
#define GRID_GEMM(M) dim3((unsigned)(((M) + 127) / 128)), dim3(256)

extern "C" void kernel_launch(void* const* d_in, const int* in_sizes, int n_in,
                              void* d_out, int out_size, void* d_ws, size_t ws_size,
                              hipStream_t stream) {
  // ---- locate top-level inputs (insertion order vs sorted-key flatten) ----
  int i_xs, i_xt, i_ea, i_ei, p0;
  if (in_sizes[0] == NSN * 6) { i_xs = 0; i_xt = 1; i_ea = 2; i_ei = 3; p0 = 4; }
  else { i_ea = 0; i_ei = 1; p0 = 2; i_xs = n_in - 2; i_xt = n_in - 1; }
  const float* x_s       = (const float*)d_in[i_xs];
  const float* x_t       = (const float*)d_in[i_xt];
  const float* edge_attr = (const float*)d_in[i_ea];
  const int*   ei        = (const int*)d_in[i_ei];
  const int* con_idx = ei;        // edge_index row 0
  const int* var_idx = ei + EN;   // edge_index row 1

  // ---- walk params pytree (jax sorted dict keys) ----
  int c = p0;
  ConL con[3];
  for (int i = 0; i < 3; ++i) {
    con[i].edge = nxtBlk(d_in, c); con[i].eps = nxt(d_in, c);
    con[i].joint = nxtBlk(d_in, c); con[i].mlp = nxtBlk(d_in, c);
  }
  Lin encc1 = nxtLin(d_in, c), encc2 = nxtLin(d_in, c);
  Lin encv1 = nxtLin(d_in, c), encv2 = nxtLin(d_in, c);
  VarL var[3];
  for (int i = 0; i < 3; ++i) {
    var[i].ass1 = nxtLin(d_in, c); var[i].ass2 = nxtLin(d_in, c);
    var[i].edge = nxtBlk(d_in, c); var[i].eps = nxt(d_in, c);
    var[i].joint = nxtBlk(d_in, c); var[i].mlp = nxtBlk(d_in, c);
  }

  // ---- workspace layout (~231 MB) ----
  float* ws   = (float*)d_ws;
  float* tmpA = ws;                              // 50000*128
  float* tmpB = tmpA + (size_t)NSN * DH;         // 50000*128
  float* agg  = tmpB + (size_t)NSN * DH;         // 50000*128
  float* assb = agg  + (size_t)NSN * DH;         // 50048
  float* wsum = assb + 50048;                    // 16384
  float* sum1 = wsum + 16384;                    // 4x128 stats + 4x128 affine
  float* sq1  = sum1 + 128;
  float* sum2 = sq1  + 128;
  float* sq2  = sum2 + 128;
  float* sc1  = sq2  + 128; float* sh1 = sc1 + 128;
  float* sc2  = sh1  + 128; float* sh2 = sc2 + 128;
  unsigned short* eebuf = (unsigned short*)(sh2 + 128);  // E*128 bf16

  float* out_con = (float*)d_out;                 // [NS, 512] JK-cat, col 128*i
  float* out_var = out_con + (size_t)NSN * 512;   // [NT, 512]

  // ---- encoders: Linear(relu) then Linear (no act), straight into d_out ----
  lin_small_relu<<<GRID_EW((long long)NSN * 128), 0, stream>>>(x_s, NSN, 6, encc1.W, encc1.b, tmpA);
  gemm128_wmma<0,false,false,false,false><<<GRID_GEMM(NSN), 0, stream>>>(
      NSN, tmpA, 128, nullptr, nullptr, nullptr, encc2.W, encc2.b,
      nullptr, nullptr, out_con, 512, nullptr, nullptr, nullptr);
  lin_small_relu<<<GRID_EW((long long)NTN * 128), 0, stream>>>(x_t, NTN, 9, encv1.W, encv1.b, tmpA);
  gemm128_wmma<0,false,false,false,false><<<GRID_GEMM(NTN), 0, stream>>>(
      NTN, tmpA, 128, nullptr, nullptr, nullptr, encv2.W, encv2.b,
      nullptr, nullptr, out_var, 512, nullptr, nullptr, nullptr);

  for (int i = 0; i < 3; ++i) {
    const VarL& v = var[i]; const ConL& cl = con[i];
    const float* conprev = out_con + 128 * i;
    const float* varprev = out_var + 128 * i;
    float* connew = out_con + 128 * (i + 1);
    float* varnew = out_var + 128 * (i + 1);

    // ================= variables -> constraints =================
    // assignment head: relu(xv@W)@w2 -> sigmoid
    gemm128_wmma<0,true,false,false,false><<<GRID_GEMM(NTN), 0, stream>>>(
        NTN, varprev, 512, nullptr, nullptr, nullptr, v.ass1.W, v.ass1.b,
        nullptr, nullptr, tmpA, 128, nullptr, nullptr, nullptr);
    ass2_kernel<<<dim3((NTN + 7) / 8), dim3(256), 0, stream>>>(tmpA, NTN, v.ass2.W, v.ass2.b, assb);
    // joint block: K=129 -> K=128 GEMM + rank-1(ass) epilogue; l2 with fused stats
    zero_kernel<<<dim3(2), dim3(256), 0, stream>>>(sum1, 512);
    gemm128_wmma<0,true,true,false,false><<<GRID_GEMM(NTN), 0, stream>>>(
        NTN, varprev, 512, nullptr, nullptr, nullptr, v.joint.l1.W, v.joint.l1.b,
        assb, v.joint.l1.W + 128 * 128, tmpB, 128, nullptr, nullptr, nullptr);
    gemm128_wmma<0,true,false,true,false><<<GRID_GEMM(NTN), 0, stream>>>(
        NTN, tmpB, 128, nullptr, nullptr, nullptr, v.joint.l2.W, v.joint.l2.b,
        nullptr, nullptr, tmpA, 128, nullptr, sum1, sq1);
    // edge block: layer-1 generated in A-tile loader, bf16 output + fused stats
    gemm128_wmma<1,true,false,true,true><<<GRID_GEMM(EN), 0, stream>>>(
        EN, nullptr, 0, edge_attr, v.edge.l1.W, v.edge.l1.b, v.edge.l2.W, v.edge.l2.b,
        nullptr, nullptr, nullptr, 128, eebuf, sum2, sq2);
    bn_finalize<<<dim3(1), dim3(128), 0, stream>>>(sum1, sq1, (float)NTN, v.joint.g, v.joint.beta, sc1, sh1);
    bn_finalize<<<dim3(1), dim3(128), 0, stream>>>(sum2, sq2, (float)EN,  v.edge.g,  v.edge.beta,  sc2, sh2);
    // message passing: gather(var) + BN-fused add + relu + atomic segment-sum(con)
    zero_kernel<<<GRID_EW((long long)NSN * 128), 0, stream>>>(agg, (long long)NSN * 128);
    scatter_kernel<<<GRID_EW((long long)EN * 32), 0, stream>>>(
        tmpA, eebuf, var_idx, con_idx, EN, sc1, sh1, sc2, sh2, agg);
    // GIN update: relu(BN(mlp((1+eps)*xc + agg)))
    gin_kernel<<<GRID_EW((long long)NSN * 32), 0, stream>>>(conprev, 512, agg, v.eps, NSN, tmpB);
    zero_kernel<<<dim3(1), dim3(256), 0, stream>>>(sum1, 256);
    gemm128_wmma<0,true,false,false,false><<<GRID_GEMM(NSN), 0, stream>>>(
        NSN, tmpB, 128, nullptr, nullptr, nullptr, v.mlp.l1.W, v.mlp.l1.b,
        nullptr, nullptr, tmpA, 128, nullptr, nullptr, nullptr);
    gemm128_wmma<0,true,false,true,false><<<GRID_GEMM(NSN), 0, stream>>>(
        NSN, tmpA, 128, nullptr, nullptr, nullptr, v.mlp.l2.W, v.mlp.l2.b,
        nullptr, nullptr, tmpB, 128, nullptr, sum1, sq1);
    bn_finalize<<<dim3(1), dim3(128), 0, stream>>>(sum1, sq1, (float)NSN, v.mlp.g, v.mlp.beta, sc1, sh1);
    bn_relu_kernel<<<GRID_EW((long long)NSN * 32), 0, stream>>>(tmpB, NSN, sc1, sh1, connew, 512);

    // ================= constraints -> variables =================
    // concat([nc,nc]) @ W  ==  nc @ (W_top + W_bot)
    wsum_kernel<<<dim3(64), dim3(256), 0, stream>>>(cl.joint.l1.W, wsum);
    zero_kernel<<<dim3(2), dim3(256), 0, stream>>>(sum1, 512);
    gemm128_wmma<0,true,false,false,false><<<GRID_GEMM(NSN), 0, stream>>>(
        NSN, connew, 512, nullptr, nullptr, nullptr, wsum, cl.joint.l1.b,
        nullptr, nullptr, tmpA, 128, nullptr, nullptr, nullptr);
    gemm128_wmma<0,true,false,true,false><<<GRID_GEMM(NSN), 0, stream>>>(
        NSN, tmpA, 128, nullptr, nullptr, nullptr, cl.joint.l2.W, cl.joint.l2.b,
        nullptr, nullptr, tmpB, 128, nullptr, sum1, sq1);
    gemm128_wmma<1,true,false,true,true><<<GRID_GEMM(EN), 0, stream>>>(
        EN, nullptr, 0, edge_attr, cl.edge.l1.W, cl.edge.l1.b, cl.edge.l2.W, cl.edge.l2.b,
        nullptr, nullptr, nullptr, 128, eebuf, sum2, sq2);
    bn_finalize<<<dim3(1), dim3(128), 0, stream>>>(sum1, sq1, (float)NSN, cl.joint.g, cl.joint.beta, sc1, sh1);
    bn_finalize<<<dim3(1), dim3(128), 0, stream>>>(sum2, sq2, (float)EN,  cl.edge.g,  cl.edge.beta,  sc2, sh2);
    zero_kernel<<<GRID_EW((long long)NTN * 128), 0, stream>>>(agg, (long long)NTN * 128);
    scatter_kernel<<<GRID_EW((long long)EN * 32), 0, stream>>>(
        tmpB, eebuf, con_idx, var_idx, EN, sc1, sh1, sc2, sh2, agg);
    gin_kernel<<<GRID_EW((long long)NTN * 32), 0, stream>>>(varprev, 512, agg, cl.eps, NTN, tmpA);
    zero_kernel<<<dim3(1), dim3(256), 0, stream>>>(sum1, 256);
    gemm128_wmma<0,true,false,false,false><<<GRID_GEMM(NTN), 0, stream>>>(
        NTN, tmpA, 128, nullptr, nullptr, nullptr, cl.mlp.l1.W, cl.mlp.l1.b,
        nullptr, nullptr, tmpB, 128, nullptr, nullptr, nullptr);
    gemm128_wmma<0,true,false,true,false><<<GRID_GEMM(NTN), 0, stream>>>(
        NTN, tmpB, 128, nullptr, nullptr, nullptr, cl.mlp.l2.W, cl.mlp.l2.b,
        nullptr, nullptr, tmpA, 128, nullptr, sum1, sq1);
    bn_finalize<<<dim3(1), dim3(128), 0, stream>>>(sum1, sq1, (float)NTN, cl.mlp.g, cl.mlp.beta, sc1, sh1);
    bn_relu_kernel<<<GRID_EW((long long)NTN * 32), 0, stream>>>(tmpA, NTN, sc1, sh1, varnew, 512);
  }
  (void)out_size; (void)ws_size;
}
